// MoE_7267084665536
// MI455X (gfx1250) — matrix operand order repaired and checked by reading, not compiled, not run on hardware
//
#include <hip/hip_runtime.h>
#include <hip/hip_bf16.h>
#include <math.h>

// ---------------- problem constants ----------------
#define H_DIM 1024
#define FF_DIM 4096
#define NEXP 8
#define NTOK 4096   // 2 * 2048

typedef _Float16 half_t;
typedef __attribute__((ext_vector_type(16))) _Float16 v16h;
typedef __attribute__((ext_vector_type(8)))  _Float16 v8h;
typedef __attribute__((ext_vector_type(2)))  _Float16 v2h;
typedef __attribute__((ext_vector_type(8)))  float    v8f;
typedef __attribute__((ext_vector_type(4)))  float    v4f;
typedef __attribute__((ext_vector_type(2)))  float    v2f;

// ---------------- GEMM tiling ----------------
#define TBM 128    // token rows per block (4 wave-rows * 32)
#define TBN 64     // output cols per block (2 wave-cols * 32)
#define TBK 64     // K per LDS stage == 2 WMMA k-slices
#define KST 72     // LDS K-stride in halves (144B: 16B-aligned frags, low bank conflict)

// ------------- helpers -------------
__device__ __forceinline__ float gelu_tanh(float v) {
    const float c0 = 0.7978845608028654f;  // sqrt(2/pi)
    const float c1 = 0.044715f;
    float u = c0 * (v + c1 * v * v * v);
    return 0.5f * v * (1.0f + tanhf(u));
}

// A fragment (16x32 f16, ISA 7.12.2 "16-bit A-Matrix 16x32"):
// lane<16 row=lane: elems 0..7 = K 0..7, elems 8..15 = K 16..23
// lane>=16 row=lane-16: elems 0..7 = K 8..15, elems 8..15 = K 24..31
__device__ __forceinline__ v16h load_frag_a(const half_t* __restrict__ p, int hs) {
    v8h lo = *(const v8h*)(p + hs * 8);
    v8h hi = *(const v8h*)(p + 16 + hs * 8);
    v16h f;
#pragma unroll
    for (int i = 0; i < 8; ++i) { f[i] = lo[i]; f[i + 8] = hi[i]; }
    return f;
}

// B fragment (32x16 f16, same K order as A fragment):
// lane<16 col=lane: elems 0..7 = K 0..7, elems 8..15 = K 16..23 ; lane>=16: elems 0..7 = K 8..15, elems 8..15 = K 24..31
// Stored transposed in LDS as [n][k].
__device__ __forceinline__ v16h load_frag_b(const half_t* __restrict__ p, int hs) {
    v8h lo = *(const v8h*)(p + hs * 8);
    v8h hi = *(const v8h*)(p + 16 + hs * 8);
    v16h f;
#pragma unroll
    for (int i = 0; i < 8; ++i) { f[i] = lo[i]; f[i + 8] = hi[i]; }
    return f;
}

// ---------------- router: logits, softmax, top-2, combine ----------------
__global__ __launch_bounds__(256)
void moe_router(const float* __restrict__ x, const float* __restrict__ Wr,
                float* __restrict__ combine, float* __restrict__ logits_out,
                int* __restrict__ sel_out) {
    const int lane = threadIdx.x & 31;
    const int t = blockIdx.x * 8 + (threadIdx.x >> 5);
    const float* xr = x + (size_t)t * H_DIM;

    float s[NEXP];
#pragma unroll
    for (int e = 0; e < NEXP; ++e) s[e] = 0.f;

    for (int h = lane; h < H_DIM; h += 32) {
        float xv = xr[h];
        v4f w0 = *(const v4f*)(Wr + h * NEXP);
        v4f w1 = *(const v4f*)(Wr + h * NEXP + 4);
        s[0] += xv * w0.x; s[1] += xv * w0.y; s[2] += xv * w0.z; s[3] += xv * w0.w;
        s[4] += xv * w1.x; s[5] += xv * w1.y; s[6] += xv * w1.z; s[7] += xv * w1.w;
    }
#pragma unroll
    for (int off = 16; off > 0; off >>= 1) {
#pragma unroll
        for (int e = 0; e < NEXP; ++e) s[e] += __shfl_xor(s[e], off, 32);
    }
    if (lane == 0) {
        float m = s[0];
#pragma unroll
        for (int e = 1; e < NEXP; ++e) m = fmaxf(m, s[e]);
        float p[NEXP]; float sum = 0.f;
#pragma unroll
        for (int e = 0; e < NEXP; ++e) { p[e] = expf(s[e] - m); sum += p[e]; }
        float inv = 1.f / sum;
#pragma unroll
        for (int e = 0; e < NEXP; ++e) p[e] *= inv;
        // top-2, strict '>' so lowest index wins ties (matches lax.top_k)
        int i0 = 0; float b0 = p[0];
#pragma unroll
        for (int e = 1; e < NEXP; ++e) if (p[e] > b0) { b0 = p[e]; i0 = e; }
        int i1 = (i0 == 0) ? 1 : 0; float b1v = p[i1];
#pragma unroll
        for (int e = 0; e < NEXP; ++e)
            if (e != i0 && p[e] > b1v) { b1v = p[e]; i1 = e; }
#pragma unroll
        for (int e = 0; e < NEXP; ++e) {
            combine[t * NEXP + e] = (e == i0) ? b0 : ((e == i1) ? b1v : 0.f);
            logits_out[t * NEXP + e] = s[e];
        }
        sel_out[t * 2 + 0] = i0;
        sel_out[t * 2 + 1] = i1;
    }
}

// ---------------- ordered per-expert token compaction ----------------
__global__ void moe_build_lists(const float* __restrict__ combine,
                                int* __restrict__ tok, float* __restrict__ gw,
                                int* __restrict__ counts) {
    const int e = blockIdx.x;
    const int lane = threadIdx.x;
    int cnt = 0;
    for (int t0 = 0; t0 < NTOK; t0 += 32) {
        int t = t0 + lane;
        float w = combine[t * NEXP + e];
        bool sel = (w > 0.f);
        unsigned mask = (unsigned)__ballot(sel);
        int pre = __popc(mask & ((1u << lane) - 1u));
        if (sel) {
            tok[e * NTOK + cnt + pre] = t;
            gw[e * NTOK + cnt + pre] = w;
        }
        cnt += __popc(mask);
    }
    if (lane == 0) counts[e] = cnt;
}

// ---------------- GEMM1: h = gelu(gather(x) @ W1[e] + b1[e]) -> f16 ----------------
__global__ __launch_bounds__(256)
void moe_gemm1(const float* __restrict__ x, const float* __restrict__ W1,
               const float* __restrict__ b1, const int* __restrict__ token_ids,
               const int* __restrict__ counts, half_t* __restrict__ hbuf, int e) {
    __shared__ half_t As[TBM * KST];   // [m][k]
    __shared__ half_t Bs[TBN * KST];   // [n][k] (transposed W1 tile)

    const int cnt = counts[e];
    const int m0 = blockIdx.y * TBM;
    if (m0 >= cnt) return;
    const int n0 = blockIdx.x * TBN;

    const int tid = threadIdx.x;
    const int lane = tid & 31;
    const int wid = tid >> 5;
    const int wm = wid >> 1;      // 0..3 wave row
    const int wn = wid & 1;       // 0..1 wave col
    const int hs = lane >> 4;     // lane half
    const int ln = lane & 15;

    const float* W1e = W1 + (size_t)e * H_DIM * FF_DIM;
    const int* tokp = token_ids + e * NTOK;

    // ---- hoisted A gather setup: 8 fixed (row,kq) slots per thread ----
    // Rows beyond cnt are CLAMPED (not masked): a garbage A row only affects
    // its own (discarded) C row, so no guards are needed in the hot loop.
    const float* aptr[8];
    int adst[8];
#pragma unroll
    for (int i = 0; i < 8; ++i) {
        int g = tid + 256 * i;        // 0..2047
        int row = g >> 4;             // 0..127
        int kq = (g & 15) * 4;        // 0..60
        int gm = m0 + row;
        int gmc = (gm < cnt) ? gm : (cnt - 1);
        adst[i] = row * KST + kq;
        aptr[i] = x + (size_t)tokp[gmc] * H_DIM + kq;
    }

    // ---- hoisted B setup: 4 items/thread, each covers 2k x 2n ----
    size_t boff[4]; int bdst[4];
#pragma unroll
    for (int i = 0; i < 4; ++i) {
        int g = tid + 256 * i;        // 0..1023
        int kp = g >> 5;              // 0..31 (k pair)
        int n2 = (g & 31) * 2;
        boff[i] = (size_t)(2 * kp) * FF_DIM + n0 + n2;
        bdst[i] = n2 * KST + 2 * kp;
    }

    v8f acc[2][2];
#pragma unroll
    for (int mi = 0; mi < 2; ++mi)
#pragma unroll
        for (int ni = 0; ni < 2; ++ni)
#pragma unroll
            for (int r = 0; r < 8; ++r) acc[mi][ni][r] = 0.f;

    // prologue prefetch of tile k0=0
    v4f areg[8]; v2f breg0[4], breg1[4];
#pragma unroll
    for (int i = 0; i < 8; ++i) areg[i] = *(const v4f*)(aptr[i]);
#pragma unroll
    for (int i = 0; i < 4; ++i) {
        const float* bp = W1e + boff[i];
        breg0[i] = *(const v2f*)(bp);
        breg1[i] = *(const v2f*)(bp + FF_DIM);
    }

    for (int k0 = 0; k0 < H_DIM; k0 += TBK) {
        // stage current tile (registers -> LDS, fp32 -> f16)
#pragma unroll
        for (int i = 0; i < 8; ++i) {
            half_t* d = &As[adst[i]];
            d[0] = (half_t)areg[i].x; d[1] = (half_t)areg[i].y;
            d[2] = (half_t)areg[i].z; d[3] = (half_t)areg[i].w;
        }
#pragma unroll
        for (int i = 0; i < 4; ++i) {
            v2h lo; lo[0] = (half_t)breg0[i].x; lo[1] = (half_t)breg1[i].x;
            v2h hi; hi[0] = (half_t)breg0[i].y; hi[1] = (half_t)breg1[i].y;
            *(v2h*)&Bs[bdst[i]]       = lo;   // row n2,   k..k+1
            *(v2h*)&Bs[bdst[i] + KST] = hi;   // row n2+1, k..k+1
        }
        __syncthreads();

        // prefetch next tile into registers (overlaps with WMMA below)
        int kn = k0 + TBK;
        if (kn < H_DIM) {
#pragma unroll
            for (int i = 0; i < 8; ++i) areg[i] = *(const v4f*)(aptr[i] + kn);
#pragma unroll
            for (int i = 0; i < 4; ++i) {
                const float* bp = W1e + boff[i] + (size_t)kn * FF_DIM;
                breg0[i] = *(const v2f*)(bp);
                breg1[i] = *(const v2f*)(bp + FF_DIM);
            }
        }

        // two WMMA k-slices per staged tile
#pragma unroll
        for (int kk = 0; kk < 2; ++kk) {
            v16h af[2], bf[2];
#pragma unroll
            for (int mi = 0; mi < 2; ++mi)
                af[mi] = load_frag_a(&As[(wm * 32 + mi * 16 + ln) * KST + kk * 32], hs);
#pragma unroll
            for (int ni = 0; ni < 2; ++ni)
                bf[ni] = load_frag_b(&Bs[(wn * 32 + ni * 16 + ln) * KST + kk * 32], hs);
#pragma unroll
            for (int mi = 0; mi < 2; ++mi)
#pragma unroll
                for (int ni = 0; ni < 2; ++ni)
                    acc[mi][ni] = __builtin_amdgcn_wmma_f32_16x16x32_f16(
                        false, af[mi], false, bf[ni], (short)0, acc[mi][ni], false, false);
        }
        __syncthreads();
    }

    // epilogue: gelu(acc + b1) -> f16 h buffer
    const float* b1e = b1 + e * FF_DIM;
#pragma unroll
    for (int mi = 0; mi < 2; ++mi) {
#pragma unroll
        for (int ni = 0; ni < 2; ++ni) {
            int col = n0 + wn * 32 + ni * 16 + ln;
            float bias = b1e[col];
#pragma unroll
            for (int r = 0; r < 8; ++r) {
                int row = wm * 32 + mi * 16 + r + hs * 8;
                int gm = m0 + row;
                if (gm < cnt) {
                    float v = acc[mi][ni][r] + bias;
                    hbuf[(size_t)gm * FF_DIM + col] = (half_t)gelu_tanh(v);
                }
            }
        }
    }
}

// ---------------- GEMM2: out[t] += w * (h @ W2[e] + b2[e]) ----------------
__global__ __launch_bounds__(256)
void moe_gemm2(const half_t* __restrict__ hbuf, const float* __restrict__ W2,
               const float* __restrict__ b2, const int* __restrict__ token_ids,
               const float* __restrict__ gw, const int* __restrict__ counts,
               float* __restrict__ out, int e) {
    __shared__ half_t As[TBM * KST];
    __shared__ half_t Bs[TBN * KST];

    const int cnt = counts[e];
    const int m0 = blockIdx.y * TBM;
    if (m0 >= cnt) return;
    const int n0 = blockIdx.x * TBN;

    const int tid = threadIdx.x;
    const int lane = tid & 31;
    const int wid = tid >> 5;
    const int wm = wid >> 1;
    const int wn = wid & 1;
    const int hs = lane >> 4;
    const int ln = lane & 15;

    const float* W2e = W2 + (size_t)e * FF_DIM * H_DIM;
    const int* tokp = token_ids + e * NTOK;
    const float* gwe = gw + e * NTOK;

    // ---- hoisted A setup: 4 fixed (row,k8) slots per thread, f16 rows ----
    // Out-of-range rows clamped (their C rows are discarded in the epilogue).
    const half_t* aptr[4];
    int adst[4];
#pragma unroll
    for (int i = 0; i < 4; ++i) {
        int g = tid + 256 * i;        // 0..1023
        int row = g >> 3;             // 0..127
        int k8 = (g & 7) * 8;         // 0..56
        int gm = m0 + row;
        int gmc = (gm < cnt) ? gm : (cnt - 1);
        adst[i] = row * KST + k8;
        aptr[i] = hbuf + (size_t)gmc * FF_DIM + k8;
    }

    // ---- hoisted B setup ----
    size_t boff[4]; int bdst[4];
#pragma unroll
    for (int i = 0; i < 4; ++i) {
        int g = tid + 256 * i;
        int kp = g >> 5;
        int n2 = (g & 31) * 2;
        boff[i] = (size_t)(2 * kp) * H_DIM + n0 + n2;
        bdst[i] = n2 * KST + 2 * kp;
    }

    v8f acc[2][2];
#pragma unroll
    for (int mi = 0; mi < 2; ++mi)
#pragma unroll
        for (int ni = 0; ni < 2; ++ni)
#pragma unroll
            for (int r = 0; r < 8; ++r) acc[mi][ni][r] = 0.f;

    // prologue prefetch of tile k0=0
    v8h areg[4]; v2f breg0[4], breg1[4];
#pragma unroll
    for (int i = 0; i < 4; ++i) areg[i] = *(const v8h*)(aptr[i]);
#pragma unroll
    for (int i = 0; i < 4; ++i) {
        const float* bp = W2e + boff[i];
        breg0[i] = *(const v2f*)(bp);
        breg1[i] = *(const v2f*)(bp + H_DIM);
    }

    for (int k0 = 0; k0 < FF_DIM; k0 += TBK) {
        // stage current tile
#pragma unroll
        for (int i = 0; i < 4; ++i) *(v8h*)&As[adst[i]] = areg[i];
#pragma unroll
        for (int i = 0; i < 4; ++i) {
            v2h lo; lo[0] = (half_t)breg0[i].x; lo[1] = (half_t)breg1[i].x;
            v2h hi; hi[0] = (half_t)breg0[i].y; hi[1] = (half_t)breg1[i].y;
            *(v2h*)&Bs[bdst[i]]       = lo;
            *(v2h*)&Bs[bdst[i] + KST] = hi;
        }
        __syncthreads();

        // prefetch next tile
        int kn = k0 + TBK;
        if (kn < FF_DIM) {
#pragma unroll
            for (int i = 0; i < 4; ++i) areg[i] = *(const v8h*)(aptr[i] + kn);
#pragma unroll
            for (int i = 0; i < 4; ++i) {
                const float* bp = W2e + boff[i] + (size_t)kn * H_DIM;
                breg0[i] = *(const v2f*)(bp);
                breg1[i] = *(const v2f*)(bp + H_DIM);
            }
        }

#pragma unroll
        for (int kk = 0; kk < 2; ++kk) {
            v16h af[2], bf[2];
#pragma unroll
            for (int mi = 0; mi < 2; ++mi)
                af[mi] = load_frag_a(&As[(wm * 32 + mi * 16 + ln) * KST + kk * 32], hs);
#pragma unroll
            for (int ni = 0; ni < 2; ++ni)
                bf[ni] = load_frag_b(&Bs[(wn * 32 + ni * 16 + ln) * KST + kk * 32], hs);
#pragma unroll
            for (int mi = 0; mi < 2; ++mi)
#pragma unroll
                for (int ni = 0; ni < 2; ++ni)
                    acc[mi][ni] = __builtin_amdgcn_wmma_f32_16x16x32_f16(
                        false, af[mi], false, bf[ni], (short)0, acc[mi][ni], false, false);
        }
        __syncthreads();
    }

    // epilogue: out[t, col] += w_t * (acc + b2[col])
    // Safe without atomics: experts are processed by sequential kernel launches,
    // and within one expert each (token, col) is produced exactly once.
    const float* b2e = b2 + e * H_DIM;
#pragma unroll
    for (int mi = 0; mi < 2; ++mi) {
#pragma unroll
        for (int ni = 0; ni < 2; ++ni) {
            int col = n0 + wn * 32 + ni * 16 + ln;
            float bias = b2e[col];
#pragma unroll
            for (int r = 0; r < 8; ++r) {
                int row = wm * 32 + mi * 16 + r + hs * 8;
                int gm = m0 + row;
                if (gm < cnt) {
                    int t = tokp[gm];
                    float w = gwe[gm];
                    float y = acc[mi][ni][r] + bias;
                    float* o = out + (size_t)t * H_DIM + col;
                    *o = *o + w * y;
                }
            }
        }
    }
}

// ---------------- launcher ----------------
extern "C" void kernel_launch(void* const* d_in, const int* in_sizes, int n_in,
                              void* d_out, int out_size, void* d_ws, size_t ws_size,
                              hipStream_t stream) {
    (void)in_sizes; (void)n_in; (void)out_size; (void)ws_size;
    const float* x  = (const float*)d_in[0];
    const float* Wr = (const float*)d_in[1];
    const float* W1 = (const float*)d_in[2];
    const float* b1 = (const float*)d_in[3];
    const float* W2 = (const float*)d_in[4];
    const float* b2 = (const float*)d_in[5];
    float* out = (float*)d_out;

    // workspace layout (needs ~32.4 MB)
    char* ws = (char*)d_ws;
    float*  combine = (float*)(ws);                 // 4096*8 f32   = 128 KB
    float*  gw      = (float*)(ws + 131072);        // 8*4096 f32   = 128 KB
    int*    tok     = (int*)  (ws + 262144);        // 8*4096 i32   = 128 KB
    int*    counts  = (int*)  (ws + 393216);        // 8 i32 (padded to 512B)
    half_t* hbuf    = (half_t*)(ws + 393728);       // 4096*4096 f16 = 32 MB

    float* logits_out = out + (size_t)NTOK * H_DIM;                  // [4096,8] f32
    int*   sel_out    = (int*)out + ((size_t)NTOK * H_DIM + NTOK * NEXP); // [4096,2] i32

    // zero the accumulated-output region (graph-capturable)
    hipMemsetAsync(out, 0, (size_t)NTOK * H_DIM * sizeof(float), stream);

    moe_router<<<NTOK / 8, 256, 0, stream>>>(x, Wr, combine, logits_out, sel_out);
    moe_build_lists<<<NEXP, 32, 0, stream>>>(combine, tok, gw, counts);

    for (int e = 0; e < NEXP; ++e) {
        moe_gemm1<<<dim3(FF_DIM / TBN, NTOK / TBM), 256, 0, stream>>>(
            x, W1, b1, tok, counts, hbuf, e);
        moe_gemm2<<<dim3(H_DIM / TBN, NTOK / TBM), 256, 0, stream>>>(
            hbuf, W2, b2, tok, gw, counts, out, e);
    }
}
